// LigerCosineSimilarityLoss_50714973831635
// MI455X (gfx1250) — matrix operand order repaired and checked by reading, not compiled, not run on hardware
//
#include <hip/hip_runtime.h>

// ---------------------------------------------------------------------------
// Fused distillation loss for MI455X (gfx1250, wave32, WMMA bf16).
//   s_logits = student @ W_s^T   [2048 x 32000], K=2048
//   t_logits = teacher @ W_t^T   [2048 x 32000], K=4096
// Pass 0: one-shot f32 -> bf16 conversion of all operands into scratch.
// Pass 1: tiled WMMA GEMM, double-buffered LDS, software-pipelined so
//         global loads (next tile) -> frag ds_loads + WMMA (current tile)
//         -> ds_stores (next tile) -> barrier. Each 16x16 D-tile is folded
//         into per-row online stats; logits never hit memory.
// Pass 2: combine per-chunk stats -> scalar loss.
// ---------------------------------------------------------------------------

typedef __attribute__((ext_vector_type(16))) __bf16 v16bf;
typedef __attribute__((ext_vector_type(8)))  float  v8f;

#define HS      2048            // student hidden
#define HT      4096            // teacher hidden
#define VDIM    32000           // vocab
#define BTDIM   2048            // rows (B*T)
#define BM      64              // rows per block (4 waves in M)
#define BN      128             // cols per inner iteration (2 wave groups x 64)
#define CHUNK   640             // cols per block (vocab chunk)
#define NCHUNK  (VDIM / CHUNK)  // 50
#define NITER   (CHUNK / BN)    // 5
#define LDSTRIDE 40             // bf16 elems per LDS row (32 + 8 pad, 16B-aligned)

union Frag { uint4 u[2]; v16bf bf; };

// round-to-nearest-even f32 -> bf16, packed pair (only used in pass 0)
__device__ __forceinline__ unsigned f2bf2(float lo, float hi) {
  unsigned a = __float_as_uint(lo); a = (a + 0x7FFFu + ((a >> 16) & 1u)) >> 16;
  unsigned b = __float_as_uint(hi); b = (b + 0x7FFFu + ((b >> 16) & 1u)) >> 16;
  return a | (b << 16);
}

// ---------------- Pass 0: bulk f32 -> bf16 conversion ----------------------
__global__ void __launch_bounds__(256)
convert_bf16(const float* __restrict__ src, unsigned short* __restrict__ dst,
             long long n8) {
  const long long i = (long long)blockIdx.x * 256 + threadIdx.x;
  if (i >= n8) return;
  const float4* p = (const float4*)src + i * 2;
  const float4 f0 = p[0];
  const float4 f1 = p[1];
  uint4 o;
  o.x = f2bf2(f0.x, f0.y); o.y = f2bf2(f0.z, f0.w);
  o.z = f2bf2(f1.x, f1.y); o.w = f2bf2(f1.z, f1.w);
  ((uint4*)dst)[i] = o;
}

// ---------------- Pass 1: GEMM + online stats ------------------------------
// Fragment reads + 4 WMMAs for the current LDS buffers.
__device__ __forceinline__ void mma_step(const unsigned short* __restrict__ Acur,
                                         const unsigned short* __restrict__ Bcur,
                                         int aoff0, int aoff1,
                                         int wn, int l15, int hi, v8f (&acc)[4]) {
  Frag a;
  a.u[0] = *(const uint4*)&Acur[aoff0];
  a.u[1] = *(const uint4*)&Acur[aoff1];
#pragma unroll
  for (int j = 0; j < 4; ++j) {
    const int boff = (wn * 64 + j * 16 + l15) * LDSTRIDE + hi * 8;
    Frag b;
    b.u[0] = *(const uint4*)&Bcur[boff];
    b.u[1] = *(const uint4*)&Bcur[boff + 16];
    acc[j] = __builtin_amdgcn_wmma_f32_16x16x32_bf16(
        false, a.bf, false, b.bf, (short)0, acc[j], false, false);
  }
}

// One K-pass of D += A x B, double-buffered LDS, explicit register pipeline.
__device__ __forceinline__ void gemm_pass(const unsigned short* __restrict__ Xbf,
                                          const unsigned short* __restrict__ Wbf,
                                          int H, int rowBase, int nbase,
                                          unsigned short* A0, unsigned short* A1,
                                          unsigned short* B0, unsigned short* B1,
                                          int tid, int aoff0, int aoff1,
                                          int wn, int l15, int hi, v8f (&acc)[4]) {
  // Per-thread staging addresses: A = 1x16B, B = 2x16B.
  const int r  = tid >> 2;
  const int kk = (tid & 3) * 8;
  const unsigned short* gA  = Xbf + (size_t)(rowBase + r)      * H + kk;
  const unsigned short* gB0 = Wbf + (size_t)(nbase   + r)      * H + kk;
  const unsigned short* gB1 = Wbf + (size_t)(nbase   + r + 64) * H + kk;
  const int sA  = r * LDSTRIDE + kk;
  const int sB0 = r * LDSTRIDE + kk;
  const int sB1 = (r + 64) * LDSTRIDE + kk;

  // Prologue: stage k = 0 into buffer 0.
  {
    const uint4 va  = *(const uint4*)gA;
    const uint4 vb0 = *(const uint4*)gB0;
    const uint4 vb1 = *(const uint4*)gB1;
    *(uint4*)&A0[sA]  = va;
    *(uint4*)&B0[sB0] = vb0;
    *(uint4*)&B0[sB1] = vb1;
  }
  __syncthreads();

  // Steady state: k-step i consumes buf(i&1), stages i+1 into buf(~i&1).
  for (int k0 = 0; k0 < H - 32; k0 += 32) {
    unsigned short* Acur = (k0 & 32) ? A1 : A0;
    unsigned short* Bcur = (k0 & 32) ? B1 : B0;
    unsigned short* Anxt = (k0 & 32) ? A0 : A1;
    unsigned short* Bnxt = (k0 & 32) ? B0 : B1;

    // (1) long-latency global loads for the NEXT tile first
    const uint4 va  = *(const uint4*)(gA  + k0 + 32);
    const uint4 vb0 = *(const uint4*)(gB0 + k0 + 32);
    const uint4 vb1 = *(const uint4*)(gB1 + k0 + 32);

    // (2) current tile: frag ds_loads + WMMAs (no pending ds_stores to wait on)
    mma_step(Acur, Bcur, aoff0, aoff1, wn, l15, hi, acc);

    // (3) stage next tile (waits on loadcnt only, after WMMAs issued)
    *(uint4*)&Anxt[sA]  = va;
    *(uint4*)&Bnxt[sB0] = vb0;
    *(uint4*)&Bnxt[sB1] = vb1;

    // (4) one barrier per k-step
    __syncthreads();
  }

  // Epilogue: consume the final tile.
  {
    const int last = H - 32;
    unsigned short* Acur = (last & 32) ? A1 : A0;
    unsigned short* Bcur = (last & 32) ? B1 : B0;
    mma_step(Acur, Bcur, aoff0, aoff1, wn, l15, hi, acc);
  }
  __syncthreads();   // buffers reused by the next gemm_pass
}

__global__ void __launch_bounds__(256)
distill_gemm_stats(const unsigned short* __restrict__ sbf,
                   const unsigned short* __restrict__ wsbf,
                   const unsigned short* __restrict__ tbf,
                   const unsigned short* __restrict__ wtbf,
                   const int* __restrict__ target, float* __restrict__ stats) {
  __shared__ unsigned short Alds[2][BM * LDSTRIDE];   // 2 x  5 KB
  __shared__ unsigned short Blds[2][BN * LDSTRIDE];   // 2 x 10 KB

  const int tid  = threadIdx.x;
  const int lane = tid & 31;
  const int wave = tid >> 5;
  const int wm   = wave & 3;          // M-tile 0..3
  const int wn   = wave >> 2;         // N group 0..1
  const int l15  = lane & 15;
  const int hi   = lane >> 4;

  const int rowBase   = blockIdx.x * BM;
  const int chunk     = blockIdx.y;
  const int chunkBase = chunk * CHUNK;

  // D-tile layout: element i of this lane is row (hi*8 + i), col (lane&15).
  const int rbase = rowBase + wm * 16 + hi * 8;
  int tg[8];
#pragma unroll
  for (int i = 0; i < 8; ++i) tg[i] = target[rbase + i];

  float ss[8], tt[8], st[8], mx[8], se[8], sg[8];
#pragma unroll
  for (int i = 0; i < 8; ++i) { ss[i]=tt[i]=st[i]=se[i]=sg[i]=0.f; mx[i]=-3.0e38f; }

  // 16-bit A frag: lane holds K = [hi*8, hi*8+8) and [16+hi*8, 16+hi*8+8)
  const int aoff0 = (wm * 16 + l15) * LDSTRIDE + hi * 8;
  const int aoff1 = aoff0 + 16;

  const v8f vzero = {0.f, 0.f, 0.f, 0.f, 0.f, 0.f, 0.f, 0.f};

  for (int it = 0; it < NITER; ++it) {
    const int nbase = chunkBase + it * BN;
    v8f sA[4], tA[4];
#pragma unroll
    for (int j = 0; j < 4; ++j) { sA[j] = vzero; tA[j] = vzero; }

    gemm_pass(sbf, wsbf, HS, rowBase, nbase, Alds[0], Alds[1], Blds[0], Blds[1],
              tid, aoff0, aoff1, wn, l15, hi, sA);
    gemm_pass(tbf, wtbf, HT, rowBase, nbase, Alds[0], Alds[1], Blds[0], Blds[1],
              tid, aoff0, aoff1, wn, l15, hi, tA);

    // fold tiles into per-row online stats
#pragma unroll
    for (int j = 0; j < 4; ++j) {
      const int ncol = nbase + wn * 64 + j * 16 + l15;
#pragma unroll
      for (int i = 0; i < 8; ++i) {
        const float s = sA[j][i];
        const float t = tA[j][i];
        ss[i] += s * s;
        tt[i] += t * t;
        st[i] += s * t;
        const float mo = mx[i];
        const float mn = fmaxf(mo, s);
        se[i] = se[i] * __expf(mo - mn) + __expf(s - mn);
        mx[i] = mn;
        if (tg[i] == ncol) sg[i] += s;
      }
    }
  }

  // cross-lane reduction within each 16-lane half (cols 0..15 of the tile)
#pragma unroll
  for (int i = 0; i < 8; ++i) {
#pragma unroll
    for (int off = 1; off < 16; off <<= 1) {
      ss[i] += __shfl_xor(ss[i], off, 32);
      tt[i] += __shfl_xor(tt[i], off, 32);
      st[i] += __shfl_xor(st[i], off, 32);
      sg[i] += __shfl_xor(sg[i], off, 32);
      const float mo = __shfl_xor(mx[i], off, 32);
      const float so = __shfl_xor(se[i], off, 32);
      const float mn = fmaxf(mx[i], mo);
      se[i] = se[i] * __expf(mx[i] - mn) + so * __expf(mo - mn);
      mx[i] = mn;
    }
  }

  if (l15 == 0) {
    float* ws_ss = stats;
    float* ws_tt = stats + 1 * (size_t)BTDIM * NCHUNK;
    float* ws_st = stats + 2 * (size_t)BTDIM * NCHUNK;
    float* ws_mx = stats + 3 * (size_t)BTDIM * NCHUNK;
    float* ws_se = stats + 4 * (size_t)BTDIM * NCHUNK;
    float* ws_sg = stats + 5 * (size_t)BTDIM * NCHUNK;
#pragma unroll
    for (int i = 0; i < 8; ++i) {
      const int idx = (rbase + i) * NCHUNK + chunk;
      ws_ss[idx] = ss[i]; ws_tt[idx] = tt[i]; ws_st[idx] = st[i];
      ws_mx[idx] = mx[i]; ws_se[idx] = se[i]; ws_sg[idx] = sg[i];
    }
  }
}

// ---------------- Pass 2: combine chunk stats -> scalar loss ---------------
__global__ void __launch_bounds__(256)
distill_reduce(const float* __restrict__ stats, const int* __restrict__ target,
               float* __restrict__ out) {
  const int row = blockIdx.x * 256 + threadIdx.x;
  const float* ws_ss = stats;
  const float* ws_tt = stats + 1 * (size_t)BTDIM * NCHUNK;
  const float* ws_st = stats + 2 * (size_t)BTDIM * NCHUNK;
  const float* ws_mx = stats + 3 * (size_t)BTDIM * NCHUNK;
  const float* ws_se = stats + 4 * (size_t)BTDIM * NCHUNK;
  const float* ws_sg = stats + 5 * (size_t)BTDIM * NCHUNK;

  float M = -3.0e38f;
  for (int c = 0; c < NCHUNK; ++c) M = fmaxf(M, ws_mx[row * NCHUNK + c]);
  float ssT = 0.f, ttT = 0.f, stT = 0.f, seT = 0.f, sgT = 0.f;
  for (int c = 0; c < NCHUNK; ++c) {
    const int idx = row * NCHUNK + c;
    ssT += ws_ss[idx];
    ttT += ws_tt[idx];
    stT += ws_st[idx];
    seT += ws_se[idx] * __expf(ws_mx[idx] - M);
    sgT += ws_sg[idx];
  }
  const float lse  = M + __logf(seT);
  const int   tgt  = target[row];
  const float nll  = (tgt != -100) ? (lse - sgT) : 0.f;
  const float sn   = fmaxf(sqrtf(ssT), 1e-12f);
  const float tn   = fmaxf(sqrtf(ttT), 1e-12f);
  const float cosv = stT / (sn * tn);
  float contrib = (0.5f * nll + 0.5f * (1.f - cosv)) * (1.0f / (float)BTDIM);

  __shared__ float red[256];
  red[threadIdx.x] = contrib;
  __syncthreads();
  for (int s = 128; s > 0; s >>= 1) {
    if (threadIdx.x < s) red[threadIdx.x] += red[threadIdx.x + s];
    __syncthreads();
  }
  if (threadIdx.x == 0) atomicAdd(out, red[0]);
}

// ---------------------------------------------------------------------------
extern "C" void kernel_launch(void* const* d_in, const int* in_sizes, int n_in,
                              void* d_out, int out_size, void* d_ws, size_t ws_size,
                              hipStream_t stream) {
  const float* student = (const float*)d_in[0];
  const float* Ws      = (const float*)d_in[1];
  const float* teacher = (const float*)d_in[2];
  const float* Wt      = (const float*)d_in[3];
  const int*   target  = (const int*)d_in[4];
  float* out = (float*)d_out;

  // Scratch layout (bf16 copies + stats), all offsets 16B-aligned:
  //   sbf   : BTDIM*HS  bf16 (  8 MB)
  //   tbf   : BTDIM*HT  bf16 ( 16 MB)
  //   wsbf  : VDIM*HS   bf16 (128 MB)
  //   wtbf  : VDIM*HT   bf16 (256 MB)
  //   stats : 6*BTDIM*NCHUNK f32 (2.4 MB)       total ~410.4 MB
  unsigned short* sbf  = (unsigned short*)d_ws;
  unsigned short* tbf  = sbf  + (size_t)BTDIM * HS;
  unsigned short* wsbf = tbf  + (size_t)BTDIM * HT;
  unsigned short* wtbf = wsbf + (size_t)VDIM * HS;
  float*          stats = (float*)(wtbf + (size_t)VDIM * HT);

  hipMemsetAsync(d_out, 0, sizeof(float), stream);

  // Pass 0: conversions (memory-bound)
  {
    const size_t ns  = (size_t)BTDIM * HS / 8;
    const size_t nt  = (size_t)BTDIM * HT / 8;
    const size_t nws = (size_t)VDIM * HS / 8;
    const size_t nwt = (size_t)VDIM * HT / 8;
    convert_bf16<<<dim3((unsigned)((ns  + 255) / 256)), dim3(256), 0, stream>>>(student, sbf,  (long long)ns);
    convert_bf16<<<dim3((unsigned)((nt  + 255) / 256)), dim3(256), 0, stream>>>(teacher, tbf,  (long long)nt);
    convert_bf16<<<dim3((unsigned)((nws + 255) / 256)), dim3(256), 0, stream>>>(Ws,      wsbf, (long long)nws);
    convert_bf16<<<dim3((unsigned)((nwt + 255) / 256)), dim3(256), 0, stream>>>(Wt,      wtbf, (long long)nwt);
  }

  // Pass 1: WMMA GEMM + online stats. x fastest -> W-chunk stays L2-hot.
  dim3 grid1(BTDIM / BM, NCHUNK);   // (32, 50)
  distill_gemm_stats<<<grid1, dim3(256), 0, stream>>>(sbf, wsbf, tbf, wtbf,
                                                      target, stats);

  // Pass 2: scalar loss
  distill_reduce<<<dim3(BTDIM / 256), dim3(256), 0, stream>>>(stats, target, out);
}
